// MAPPORNNAgent_34196529611216
// MI455X (gfx1250) — matrix operand order repaired and checked by reading, not compiled, not run on hardware
//
#include <hip/hip_runtime.h>
#include <hip/hip_bf16.h>

#define T_STEPS 256
#define BATCH   1024
#define OBS_D   64
#define ACT_D   16
#define GD_D    512
#define H_D     128
#define H3_D    384

typedef __attribute__((ext_vector_type(16))) __bf16 v16bf;
typedef __attribute__((ext_vector_type(8)))  float  v8f;

union FragCvt { uint4 u[2]; v16bf v; };

__device__ __forceinline__ __bf16 f2bf(float f) { return (__bf16)f; }

__device__ __forceinline__ float fast_rcp(float x) {
#if __has_builtin(__builtin_amdgcn_rcpf)
  return __builtin_amdgcn_rcpf(x);
#else
  return 1.f / x;
#endif
}
__device__ __forceinline__ float fast_sigmoid(float x) {
  return fast_rcp(1.f + __expf(-x));
}
__device__ __forceinline__ float fast_tanh(float x) {
#if __has_builtin(__builtin_amdgcn_tanhf)
  return __builtin_amdgcn_tanhf(x);          // v_tanh_f32 on gfx1250
#elif __has_builtin(__builtin_amdgcn_tanh_f32)
  return __builtin_amdgcn_tanh_f32(x);
#else
  float e = __expf(2.f * x);                 // branch-free fallback
  return 1.f - 2.f * fast_rcp(e + 1.f);
#endif
}

// A-fragment (16x32 bf16) built from row-major fp32 (global). Lane L: M=L%16,
// elements 0..7 = K kb..kb+7, 8..15 = K kb+16..kb+23, kb=(L<16?0:8)+32c.
__device__ __forceinline__ v16bf a_frag_f32(const float* p /* row start + kb */) {
  v16bf a;
#pragma unroll
  for (int i = 0; i < 8; ++i) a[i] = f2bf(p[i]);
#pragma unroll
  for (int i = 0; i < 8; ++i) a[8 + i] = f2bf(p[16 + i]);
  return a;
}

// Same fragment but from a row-major bf16 array (LDS): two 16B vector loads, no cvt.
__device__ __forceinline__ v16bf a_frag_bf16(const __bf16* row, int kb) {
  FragCvt f;
  f.u[0] = *(const uint4*)(row + kb);
  f.u[1] = *(const uint4*)(row + kb + 16);
  return f.v;
}

__device__ __forceinline__ v8f wmma_bf16(v16bf a, v16bf b, v8f c) {
  return __builtin_amdgcn_wmma_f32_16x16x32_bf16(false, a, false, b, (short)0, c, false, false);
}

__device__ __forceinline__ v16bf b_frag_lds(const uint4* base, int fi) {
  FragCvt f; f.u[0] = base[fi]; f.u[1] = base[fi + 1]; return f.v;
}
__device__ __forceinline__ v16bf b_frag_glb(const uint4* base, int fi) {
  FragCvt f; f.u[0] = base[fi]; f.u[1] = base[fi + 1]; return f.v;
}

// ---------------------------------------------------------------------------
// Pack W (N,K) row-major fp32 -> bf16 WMMA B-fragments.
// B-frag: lane L holds N=nt*16+L%16, elems 0..7 = K kbase..kbase+7, 8..15 = K kbase+16..kbase+23,
// kbase=(L<16?0:8)+32c.  Layout: out[((nt*KC+c)*32+lane)*16+e]
// ---------------------------------------------------------------------------
__global__ void pack_weights(const float* __restrict__ W, __bf16* __restrict__ out,
                             int Kdim) {
  const int frag = blockIdx.x;
  const int lane = threadIdx.x;
  const int KC = Kdim >> 5;
  const int nt = frag / KC, c = frag % KC;
  const int n = nt * 16 + (lane & 15);
  const int kbase = ((lane & 16) ? 8 : 0) + 32 * c;
  __bf16* o = out + ((size_t)frag * 32 + lane) * 16;
#pragma unroll
  for (int e = 0; e < 16; ++e) o[e] = f2bf(W[(size_t)n * Kdim + kbase + e + ((e >= 8) ? 8 : 0)]);
}

struct Params {
  const float *obs, *gstate, *a_hid, *c_hid;
  const int *dones, *actions;
  const float *afc_b, *abih, *abhh, *ahd_b;
  const float *cfc_b, *cbih, *cbhh, *chd_w, *chd_b;
  const __bf16 *pk_afc, *pk_awih, *pk_awhh, *pk_ahead, *pk_cfc, *pk_cwih, *pk_cwhh;
  float *out_lp, *out_ent, *out_val;
};

// ---------------------------------------------------------------------------
// Recurrent body. One block owns 16 batch rows for all T steps; 8 waves.
// LDS: wih frags 96KB | whh frags 96KB | x bf16 4KB | h bf16 4KB | cw 512B | mask 64B
// h itself lives in registers (each wave owns columns [16*wv,16*wv+16)).
// ---------------------------------------------------------------------------
template <bool CRITIC, int KCIN>
__device__ void rnn_body(const Params p) {
  extern __shared__ char smem_raw[];
  uint4*  s_wih  = (uint4*)smem_raw;            // 96 frags * 64 uint4
  uint4*  s_whh  = s_wih + 96 * 64;
  __bf16* s_xb   = (__bf16*)(s_whh + 96 * 64);  // 16 x 128
  __bf16* s_hb   = s_xb + 16 * H_D;             // 16 x 128
  float*  s_cw   = (float*)(s_hb + 16 * H_D);   // 128 (critic head weights)
  float*  s_mask = s_cw + H_D;                  // 16

  const int b0    = blockIdx.x * 16;
  const int tid   = threadIdx.x;
  const int lane  = tid & 31;
  const int wv    = tid >> 5;
  const int col   = lane & 15;
  const int half8 = (lane & 16) ? 8 : 0;   // C-frag row offset for upper half-wave
  const int kbh   = (lane & 16) ? 8 : 0;   // A-frag K base offset

  const float*  xin    = CRITIC ? p.gstate : p.obs;
  const int     K_in   = KCIN * 32;
  const float*  hid0   = CRITIC ? p.c_hid  : p.a_hid;
  const __bf16* pk_fc  = CRITIC ? p.pk_cfc  : p.pk_afc;
  const __bf16* pk_wih = CRITIC ? p.pk_cwih : p.pk_awih;
  const __bf16* pk_whh = CRITIC ? p.pk_cwhh : p.pk_awhh;
  const float*  fc_b   = CRITIC ? p.cfc_b : p.afc_b;
  const float*  bih    = CRITIC ? p.cbih  : p.abih;
  const float*  bhh    = CRITIC ? p.cbhh  : p.abhh;

  // --- one-time: DMA weight fragments into LDS via CDNA5 async loads ---
  {
    const char* gi_ = (const char*)pk_wih;
    const char* gh_ = (const char*)pk_whh;
    for (int i = tid; i < 96 * 64; i += 256) {
      unsigned li = (unsigned)(size_t)(s_wih + i);
      unsigned lh = (unsigned)(size_t)(s_whh + i);
      unsigned long long ga = (unsigned long long)(size_t)(gi_ + i * 16);
      unsigned long long gb = (unsigned long long)(size_t)(gh_ + i * 16);
      asm volatile("global_load_async_to_lds_b128 %0, %1, off"
                   :: "v"(li), "v"(ga) : "memory");
      asm volatile("global_load_async_to_lds_b128 %0, %1, off"
                   :: "v"(lh), "v"(gb) : "memory");
    }
    asm volatile("s_wait_asynccnt 0x0" ::: "memory");
  }
  // init h: bf16 shadow in LDS (cross-wave A-frags) + f32 master in registers
  for (int i = tid; i < 16 * H_D; i += 256) {
    int r = i >> 7, c = i & 127;
    s_hb[i] = f2bf(hid0[(size_t)(b0 + r) * H_D + c]);
  }
  if (CRITIC && tid < H_D) s_cw[tid] = p.chd_w[tid];

  const int c0 = wv * 16 + col;       // this lane's gate/hidden column
  float h_reg[8];
#pragma unroll
  for (int v = 0; v < 8; ++v)
    h_reg[v] = hid0[(size_t)(b0 + v + half8) * H_D + c0];

  // loop-invariant scalars
  const float fcb   = fc_b[c0];
  const float bih_r = bih[c0],       bhh_r = bhh[c0];
  const float bih_z = bih[c0 + 128], bhh_z = bhh[c0 + 128];
  const float bih_n = bih[c0 + 256], bhh_n = bhh[c0 + 256];
  const float ahb   = CRITIC ? 0.f : p.ahd_b[col];
  const float chb   = CRITIC ? p.chd_b[0] : 0.f;
  const v8f   Z     = {0.f, 0.f, 0.f, 0.f, 0.f, 0.f, 0.f, 0.f};

  __syncthreads();

  for (int t = 0; t < T_STEPS; ++t) {
    // ---------------- phase A: read h(t-1) frags; x = tanh(in @ fc^T + b) ---
    if (tid < 16) s_mask[tid] = 1.0f - (float)p.dones[t * BATCH + b0 + tid];

    v16bf ah[4];
#pragma unroll
    for (int c = 0; c < 4; ++c) ah[c] = a_frag_bf16(s_hb + col * H_D, kbh + 32 * c);

    {
      v8f xacc = Z;
      const float* xrow = xin + (size_t)(t * BATCH + b0 + col) * K_in;
      const uint4* fcu = (const uint4*)pk_fc;
#pragma unroll
      for (int c = 0; c < KCIN; ++c) {
        v16bf af = a_frag_f32(xrow + kbh + 32 * c);
        v16bf bf = b_frag_glb(fcu, ((wv * KCIN + c) * 32 + lane) * 2);
        xacc = wmma_bf16(af, bf, xacc);
      }
      if (t + 1 < T_STEPS)
        __builtin_prefetch(xrow + (size_t)BATCH * K_in, 0, 1);
#pragma unroll
      for (int v = 0; v < 8; ++v)
        s_xb[(v + half8) * H_D + c0] = f2bf(fast_tanh(xacc[v] + fcb));
    }
    __syncthreads();  // B1: s_xb visible; all s_hb reads of h(t-1) done

    // ---------------- phase B: gi/gh WMMA, gates in registers, write h(t) ---
    {
      v16bf ax[4];
#pragma unroll
      for (int c = 0; c < 4; ++c) ax[c] = a_frag_bf16(s_xb + col * H_D, kbh + 32 * c);

      v8f air = Z, aiz = Z, ain = Z, ahr = Z, ahz = Z, ahn = Z;
#pragma unroll
      for (int c = 0; c < 4; ++c) {
        const int f_r = ((wv)      * 4 + c) * 64 + lane * 2;
        const int f_z = ((wv + 8)  * 4 + c) * 64 + lane * 2;
        const int f_n = ((wv + 16) * 4 + c) * 64 + lane * 2;
        air = wmma_bf16(ax[c], b_frag_lds(s_wih, f_r), air);
        ahr = wmma_bf16(ah[c], b_frag_lds(s_whh, f_r), ahr);
        aiz = wmma_bf16(ax[c], b_frag_lds(s_wih, f_z), aiz);
        ahz = wmma_bf16(ah[c], b_frag_lds(s_whh, f_z), ahz);
        ain = wmma_bf16(ax[c], b_frag_lds(s_wih, f_n), ain);
        ahn = wmma_bf16(ah[c], b_frag_lds(s_whh, f_n), ahn);
      }
      // masked-row identity: (m∘h)@W^T = m∘(h@W^T); r,z,n gate math per element
#pragma unroll
      for (int v = 0; v < 8; ++v) {
        const float m  = s_mask[v + half8];
        const float rg = fast_sigmoid(air[v] + bih_r + m * ahr[v] + bhh_r);
        const float z  = fast_sigmoid(aiz[v] + bih_z + m * ahz[v] + bhh_z);
        const float nn = fast_tanh(ain[v] + bih_n + rg * (m * ahn[v] + bhh_n));
        const float hv = (1.f - z) * nn + z * (m * h_reg[v]);
        h_reg[v] = hv;
        s_hb[(v + half8) * H_D + c0] = f2bf(hv);
      }
    }
    __syncthreads();  // B2: h(t) visible; all s_xb reads done

    // ---------------- phase C: heads (wave 0 only; others run ahead) --------
    if constexpr (!CRITIC) {
      if (wv == 0) {
        v8f acc = Z;
        const uint4* hdu = (const uint4*)p.pk_ahead;
#pragma unroll
        for (int c = 0; c < 4; ++c) {
          v16bf af = a_frag_bf16(s_hb + col * H_D, kbh + 32 * c);
          acc = wmma_bf16(af, b_frag_glb(hdu, (c * 32 + lane) * 2), acc);
        }
#pragma unroll
        for (int v = 0; v < 8; ++v) {
          float l = acc[v] + ahb;      // 16 logits of row v+half8, one per lane
          float m = l;
#pragma unroll
          for (int d = 1; d < 16; d <<= 1) m = fmaxf(m, __shfl_xor(m, d, 32));
          float ss = __expf(l - m);
#pragma unroll
          for (int d = 1; d < 16; d <<= 1) ss += __shfl_xor(ss, d, 32);
          float lp = l - (m + __logf(ss));
          float ec = -__expf(lp) * lp;
#pragma unroll
          for (int d = 1; d < 16; d <<= 1) ec += __shfl_xor(ec, d, 32);
          const int gidx = t * BATCH + b0 + v + half8;
          const int a = p.actions[gidx];
          if (col == 0) p.out_ent[gidx] = ec;
          if (col == a) p.out_lp[gidx] = lp;
        }
      }
    } else {
      if (wv == 0) {  // value head: 2 lanes per row, 64 FMAs each + shfl combine
        const int k0 = (lane & 16) ? 64 : 0;
        const __bf16* hr = s_hb + col * H_D + k0;
        float acc = 0.f;
#pragma unroll
        for (int k = 0; k < 64; ++k) acc += (float)hr[k] * s_cw[k0 + k];
        acc += __shfl_xor(acc, 16, 32);
        if (lane < 16) p.out_val[t * BATCH + b0 + col] = acc + chb;
      }
    }
    // no barrier needed here: next phase A only reads s_hb; s_hb is write-after-
    // read protected by B1 of the next iteration.
  }
}

__global__ __launch_bounds__(256) void rnn_scan(Params p) {
  if (blockIdx.y == 0) rnn_body<false, OBS_D / 32>(p);
  else                 rnn_body<true,  GD_D / 32>(p);
}

// ---------------------------------------------------------------------------
// Workspace layout (bf16 elements); each fragment = 512 elems (1 KB)
// ---------------------------------------------------------------------------
static constexpr size_t OFF_AFC   = 0;                       // 16 frags
static constexpr size_t OFF_AWIH  = OFF_AFC   + 16  * 512;   // 96 frags
static constexpr size_t OFF_AWHH  = OFF_AWIH  + 96  * 512;   // 96 frags
static constexpr size_t OFF_AHEAD = OFF_AWHH  + 96  * 512;   // 4 frags
static constexpr size_t OFF_CFC   = OFF_AHEAD + 4   * 512;   // 128 frags
static constexpr size_t OFF_CWIH  = OFF_CFC   + 128 * 512;   // 96 frags
static constexpr size_t OFF_CWHH  = OFF_CWIH  + 96  * 512;   // 96 frags

extern "C" void kernel_launch(void* const* d_in, const int* in_sizes, int n_in,
                              void* d_out, int out_size, void* d_ws, size_t ws_size,
                              hipStream_t stream) {
  __bf16* ws = (__bf16*)d_ws;

  Params p;
  p.obs    = (const float*)d_in[0];
  p.gstate = (const float*)d_in[1];
  p.a_hid  = (const float*)d_in[2];
  p.c_hid  = (const float*)d_in[3];
  p.dones  = (const int*)d_in[4];
  p.actions= (const int*)d_in[5];
  p.afc_b  = (const float*)d_in[7];
  p.abih   = (const float*)d_in[10];
  p.abhh   = (const float*)d_in[11];
  p.ahd_b  = (const float*)d_in[13];
  p.cfc_b  = (const float*)d_in[15];
  p.cbih   = (const float*)d_in[18];
  p.cbhh   = (const float*)d_in[19];
  p.chd_w  = (const float*)d_in[20];
  p.chd_b  = (const float*)d_in[21];
  p.pk_afc   = ws + OFF_AFC;
  p.pk_awih  = ws + OFF_AWIH;
  p.pk_awhh  = ws + OFF_AWHH;
  p.pk_ahead = ws + OFF_AHEAD;
  p.pk_cfc   = ws + OFF_CFC;
  p.pk_cwih  = ws + OFF_CWIH;
  p.pk_cwhh  = ws + OFF_CWHH;
  p.out_lp  = (float*)d_out;
  p.out_ent = p.out_lp + (size_t)T_STEPS * BATCH;
  p.out_val = p.out_lp + 2ull * T_STEPS * BATCH;

  // Phase 1: pack all GEMM weights into bf16 WMMA fragments (L2-resident)
  pack_weights<<<16,  32, 0, stream>>>((const float*)d_in[6],  ws + OFF_AFC,   OBS_D);
  pack_weights<<<96,  32, 0, stream>>>((const float*)d_in[8],  ws + OFF_AWIH,  H_D);
  pack_weights<<<96,  32, 0, stream>>>((const float*)d_in[9],  ws + OFF_AWHH,  H_D);
  pack_weights<<<4,   32, 0, stream>>>((const float*)d_in[12], ws + OFF_AHEAD, H_D);
  pack_weights<<<128, 32, 0, stream>>>((const float*)d_in[14], ws + OFF_CFC,   GD_D);
  pack_weights<<<96,  32, 0, stream>>>((const float*)d_in[16], ws + OFF_CWIH,  H_D);
  pack_weights<<<96,  32, 0, stream>>>((const float*)d_in[17], ws + OFF_CWHH,  H_D);

  // Phase 2: persistent recurrent scan — 64 batch tiles x {actor, critic}
  const int lds_bytes = 2 * 96 * 1024            // wih+whh fragments
                      + 2 * 16 * H_D * 2         // x, h bf16 shadows
                      + H_D * 4 + 16 * 4;        // critic head w + mask
  (void)hipFuncSetAttribute(reinterpret_cast<const void*>(rnn_scan),
                            hipFuncAttributeMaxDynamicSharedMemorySize, lds_bytes);
  dim3 grid(BATCH / 16, 2);
  rnn_scan<<<grid, 256, lds_bytes, stream>>>(p);
}
